// HyperConvFilter2D_45904610460143
// MI455X (gfx1250) — compile-verified, matrix operand order checked by reading, not run on hardware
//
#include <hip/hip_runtime.h>

typedef __attribute__((ext_vector_type(16))) __bf16 v16bf;
typedef __attribute__((ext_vector_type(8)))  __bf16 v8bf;
typedef __attribute__((ext_vector_type(8)))  float  v8f;

#define BANK 256
#define NADDR 64
#define OC 64
#define IC 64
#define HH 128
#define WW 128
#define BS 32
#define KTOT 576              // 9 * 64, K ordered as (kh*3+kw)*64 + i
#define WCOLS (OC * IC * 9)   // 36864
#define IPAD 72               // padded channel stride in LDS (bf16 elems, 144B = 16B-aligned)
#define ROWS 8                // output rows per block (ring-buffered input rows)

// ---------------------------------------------------------------------------
// Kernel 1: sel_w = softmax(w_addr @ address_space^T)   [32, 256]
// ---------------------------------------------------------------------------
__global__ void selw_softmax_kernel(const float* __restrict__ w_addr,
                                    const float* __restrict__ aspace,
                                    float* __restrict__ selw) {
  __shared__ float red[BANK];
  const int b = blockIdx.x, t = threadIdx.x;
  const float* wa = w_addr + b * NADDR;
  const float* as = aspace + t * NADDR;
  float logit = 0.f;
#pragma unroll 8
  for (int k = 0; k < NADDR; ++k) logit += wa[k] * as[k];
  red[t] = logit; __syncthreads();
  for (int off = BANK / 2; off > 0; off >>= 1) {
    if (t < off) red[t] = fmaxf(red[t], red[t + off]);
    __syncthreads();
  }
  const float m = red[0]; __syncthreads();
  const float e = __expf(logit - m);
  red[t] = e; __syncthreads();
  for (int off = BANK / 2; off > 0; off >>= 1) {
    if (t < off) red[t] += red[t + off];
    __syncthreads();
  }
  selw[b * BANK + t] = e / red[0];
}

// ---------------------------------------------------------------------------
// Kernel 2: bias[b,o] = softmax(b_addr[b,o] @ A^T) . b_bank
// ---------------------------------------------------------------------------
__global__ void bias_softmax_kernel(const float* __restrict__ b_addr,
                                    const float* __restrict__ b_bank,
                                    const float* __restrict__ aspace,
                                    float* __restrict__ bias) {
  __shared__ float red[BANK];
  const int bo = blockIdx.x;      // b*OC + o
  const int t  = threadIdx.x;
  const float* ba = b_addr + bo * NADDR;
  const float* as = aspace + t * NADDR;
  float logit = 0.f;
#pragma unroll 8
  for (int k = 0; k < NADDR; ++k) logit += ba[k] * as[k];
  red[t] = logit; __syncthreads();
  for (int off = BANK / 2; off > 0; off >>= 1) {
    if (t < off) red[t] = fmaxf(red[t], red[t + off]);
    __syncthreads();
  }
  const float m = red[0]; __syncthreads();
  const float e = __expf(logit - m);
  red[t] = e; __syncthreads();
  for (int off = BANK / 2; off > 0; off >>= 1) {
    if (t < off) red[t] += red[t + off];
    __syncthreads();
  }
  const float s = red[0]; __syncthreads();
  red[t] = (e / s) * b_bank[t]; __syncthreads();
  for (int off = BANK / 2; off > 0; off >>= 1) {
    if (t < off) red[t] += red[t + off];
    __syncthreads();
  }
  if (t == 0) bias[bo] = red[0];
}

// ---------------------------------------------------------------------------
// Kernel 3: W_all[b, col] = sum_n sel_w[b,n] * w_bank[n, col]
// One pass over w_bank (37.7MB read ONCE); output bf16, K = (kh*3+kw)*64 + i.
// ---------------------------------------------------------------------------
__global__ void wsynth_kernel(const float* __restrict__ selw,
                              const float* __restrict__ w_bank,
                              __bf16* __restrict__ wbf) {
  __shared__ float sel[BS * BANK];   // 32 KB
  const int t = threadIdx.x;
  for (int i = t; i < BS * BANK; i += 256) sel[i] = selw[i];
  __syncthreads();

  const int col = blockIdx.x * 256 + t;   // < WCOLS (36864 = 144 * 256)
  float acc[BS];
#pragma unroll
  for (int b = 0; b < BS; ++b) acc[b] = 0.f;

  for (int n = 0; n < BANK; ++n) {
    const float v = w_bank[(size_t)n * WCOLS + col];
#pragma unroll
    for (int b = 0; b < BS; ++b) acc[b] += sel[b * BANK + n] * v;
  }

  // w_bank col = o*576 + i*9 + (kh*3+kw)  ->  dst = o*576 + (kh*3+kw)*64 + i
  const int o   = col / (IC * 9);
  const int rem = col % (IC * 9);
  const int i   = rem / 9;
  const int kk  = rem % 9;
  const size_t dst = (size_t)o * KTOT + (size_t)kk * IC + i;
#pragma unroll
  for (int b = 0; b < BS; ++b)
    wbf[(size_t)b * WCOLS + dst] = (__bf16)acc[b];
}

// ---------------------------------------------------------------------------
// Kernel 4: per-batch 3x3 conv as implicit GEMM, V_WMMA_F32_16X16X32_BF16.
// Block = (8-row group, batch); 8 waves; wave = 16 out-ch x 64 pixels.
// 3-slot LDS ring buffer over input rows; next row prefetched into VGPRs
// during the WMMA loop, committed to LDS after compute.
// ---------------------------------------------------------------------------
__global__ __launch_bounds__(256)
void hyperconv_wmma_kernel(const float* __restrict__ x,
                           const __bf16* __restrict__ wbf,
                           const float* __restrict__ bias,
                           float* __restrict__ out) {
  __shared__ __attribute__((aligned(16))) __bf16 xs[3 * 130 * IPAD];

  const int h0  = blockIdx.x * ROWS;
  const int b   = blockIdx.y;
  const int tid = threadIdx.x;

  // per-thread fixed (w, channel-phase) for row staging: e = j*256 + tid
  const int stg_w  = tid & 127;        // 0..127
  const int stg_ih = tid >> 7;         // 0 or 1
  const float* xb = x + (size_t)b * IC * HH * WW;

  // zero halo columns of all 3 slots (never rewritten: staging uses w+1 in [1,128])
  for (int idx = tid; idx < 3 * 2 * IC; idx += 256) {
    const int r = idx / (2 * IC);
    const int j = idx % (2 * IC);
    const int w = (j < IC) ? 0 : 129;
    xs[(r * 130 + w) * IPAD + (j % IC)] = (__bf16)0.f;
  }
  // preload rows h0-1, h0, h0+1 into slots 0,1,2
  for (int idx = tid; idx < 3 * IC * WW; idx += 256) {
    const int r   = idx / (IC * WW);
    const int rm  = idx % (IC * WW);
    const int i   = rm >> 7;
    const int w   = rm & 127;
    const int hin = h0 - 1 + r;
    float v = 0.f;
    if (hin >= 0 && hin < HH) v = xb[((size_t)i * HH + hin) * WW + w];
    xs[(r * 130 + (w + 1)) * IPAD + i] = (__bf16)v;
  }
  __syncthreads();

  const int lane  = tid & 31;
  const int wid   = tid >> 5;
  const int mtile = wid & 3;    // 16 output channels
  const int nhalf = wid >> 2;   // which 64-pixel half of the row
  const int l15   = lane & 15;
  const int khalf = lane >> 4;
  const int wbase = nhalf * 64;

  const __bf16* Wrow = wbf + (size_t)b * WCOLS + (size_t)(mtile * 16 + l15) * KTOT;

  // bias per output row r of this lane (independent of h) : hoisted
  const int obase = mtile * 16 + khalf * 8;
  float bias_r[8];
#pragma unroll
  for (int r = 0; r < 8; ++r) bias_r[r] = bias[b * OC + obase + r];

  int s0 = 0;  // ring slot currently holding row (h-1)

  for (int rr = 0; rr < ROWS; ++rr) {
    const int h = h0 + rr;

    // ---- prefetch row h+2 into registers (overlaps with WMMA below)
    float pf[32];
    const bool more = (rr < ROWS - 1);
    if (more) {
      const int hin = h + 2;
#pragma unroll
      for (int j = 0; j < 32; ++j) {
        const int i = 2 * j + stg_ih;
        pf[j] = (hin < HH) ? xb[((size_t)i * HH + hin) * WW + stg_w] : 0.f;
      }
    }

    // ---- 72 WMMAs: M=16, N=4x16, K=9x64 in steps of 32
    v8f acc[4] = {};
#pragma unroll
    for (int kk = 0; kk < 9; ++kk) {
      const int kh = kk / 3;          // constant after unroll
      const int kw = kk % 3;
      const int slraw = s0 + kh;
      const int sl = (slraw >= 3) ? slraw - 3 : slraw;
      const __bf16* xrow = xs + sl * 130 * IPAD;
#pragma unroll
      for (int c = 0; c < 2; ++c) {
        const int kbase = kk * IC + c * 32;
        v8bf alo = *(const v8bf*)(Wrow + kbase + khalf * 8);
        v8bf ahi = *(const v8bf*)(Wrow + kbase + 16 + khalf * 8);
        v16bf afrag = __builtin_shufflevector(alo, ahi,
            0, 1, 2, 3, 4, 5, 6, 7, 8, 9, 10, 11, 12, 13, 14, 15);
#pragma unroll
        for (int t = 0; t < 4; ++t) {
          const int pix = wbase + t * 16 + l15 + kw;   // = w_in + 1, in [0,129]
          const __bf16* xp = xrow + pix * IPAD + c * 32 + khalf * 8;
          v8bf blo = *(const v8bf*)xp;
          v8bf bhi = *(const v8bf*)(xp + 16);
          v16bf bfrag = __builtin_shufflevector(blo, bhi,
              0, 1, 2, 3, 4, 5, 6, 7, 8, 9, 10, 11, 12, 13, 14, 15);
          acc[t] = __builtin_amdgcn_wmma_f32_16x16x32_bf16(
              false, afrag, false, bfrag, (short)0, acc[t], false, false);
        }
      }
    }

    // ---- epilogue: bias + nontemporal store (write-once, keep L2 for x/W)
#pragma unroll
    for (int r = 0; r < 8; ++r) {
      float* orow = out + (((size_t)b * OC + (obase + r)) * HH + h) * WW;
#pragma unroll
      for (int t = 0; t < 4; ++t)
        __builtin_nontemporal_store(acc[t][r] + bias_r[r],
                                    &orow[wbase + t * 16 + l15]);
    }

    // ---- rotate ring: commit prefetched row h+2 into the slot that held h-1
    __syncthreads();                       // everyone done reading slot s0
    if (more) {
#pragma unroll
      for (int j = 0; j < 32; ++j) {
        const int i = 2 * j + stg_ih;
        xs[(s0 * 130 + (stg_w + 1)) * IPAD + i] = (__bf16)pf[j];
      }
    }
    __syncthreads();                       // new row visible
    s0 = (s0 + 1 >= 3) ? 0 : s0 + 1;
  }
}

// ---------------------------------------------------------------------------
extern "C" void kernel_launch(void* const* d_in, const int* in_sizes, int n_in,
                              void* d_out, int out_size, void* d_ws, size_t ws_size,
                              hipStream_t stream) {
  const float* x      = (const float*)d_in[0];
  const float* w_addr = (const float*)d_in[1];
  const float* b_addr = (const float*)d_in[2];
  const float* w_bank = (const float*)d_in[3];
  const float* b_bank = (const float*)d_in[4];
  const float* aspace = (const float*)d_in[5];
  float* out = (float*)d_out;

  char* ws = (char*)d_ws;
  float*  selw = (float*)ws;                                   // 32*256 f32 = 32 KB
  float*  bias = (float*)(ws + BS * BANK * 4);                 // 32*64  f32 =  8 KB
  __bf16* wbf  = (__bf16*)(ws + BS * BANK * 4 + BS * OC * 4);  // 32*36864 bf16 ~ 2.3 MB

  selw_softmax_kernel<<<dim3(BS), dim3(BANK), 0, stream>>>(w_addr, aspace, selw);
  bias_softmax_kernel<<<dim3(BS * OC), dim3(BANK), 0, stream>>>(b_addr, b_bank, aspace, bias);
  wsynth_kernel<<<dim3(WCOLS / 256), dim3(256), 0, stream>>>(selw, w_bank, wbf);
  hyperconv_wmma_kernel<<<dim3(HH / ROWS, BS), dim3(256), 0, stream>>>(x, wbf, bias, out);
}
